// EncoderStage_34454227649151
// MI455X (gfx1250) — matrix operand order repaired and checked by reading, not compiled, not run on hardware
//
#include <hip/hip_runtime.h>

#define BB   4
#define CC   256
#define TT   8
#define HH   16
#define WWd  16
#define LLd  8
#define NOFF 8
#define DDim 32
#define VOX  2048          // H*W*L
#define BTn  32            // B*T
#define KKg  512           // GEMM K (feat 256 + pos 256)
#define EPSV 1e-5f

typedef __attribute__((ext_vector_type(16))) __bf16 v16bf;
typedef __attribute__((ext_vector_type(8)))  float  v8f;

static __device__ __forceinline__ unsigned short f2bf(float f) {
    union { float f; unsigned u; } x; x.f = f;
    unsigned u = x.u;
    u += 0x7fffu + ((u >> 16) & 1u);        // RNE
    return (unsigned short)(u >> 16);
}
static __device__ __forceinline__ int iclamp(int v, int lo, int hi) {
    return v < lo ? lo : (v > hi ? hi : v);
}
static __device__ __forceinline__ void wait_async0() {
#if __has_builtin(__builtin_amdgcn_s_wait_asynccnt)
    __builtin_amdgcn_s_wait_asynccnt(0);
#else
    asm volatile("s_wait_asynccnt 0x0" ::: "memory");
#endif
}

// ---------------- Kernel A: spatial mean per (bt, c) ----------------
__global__ void k_mean(const float* __restrict__ features, float* __restrict__ glob) {
    __shared__ float red[256];
    const int bi = blockIdx.x;              // bt*256 + c
    const int bt = bi >> 8, c = bi & 255;
    const int b = bt / TT, t = bt % TT;
    const float* src = features + (((size_t)(b * CC + c)) * TT + t) * VOX;
    float s = 0.f;
    for (int i = threadIdx.x; i < VOX; i += 256) s += src[i];
    red[threadIdx.x] = s; __syncthreads();
    for (int off = 128; off > 0; off >>= 1) {
        if (threadIdx.x < off) red[threadIdx.x] += red[threadIdx.x + off];
        __syncthreads();
    }
    if (threadIdx.x == 0) glob[bi] = red[0] * (1.0f / VOX);
}

// --------- Kernel B1: fold glob channels + bias: globc[bt][o] ---------
__global__ void k_globc(const float* __restrict__ Wp, const float* __restrict__ bp,
                        const float* __restrict__ glob, float* __restrict__ globc) {
    const int bt = blockIdx.x, o = threadIdx.x;
    const float* g = glob + bt * CC;
    const float* w = Wp + (size_t)o * (3 * CC) + 2 * CC;
    float s = bp[o];
    for (int c = 0; c < CC; ++c) s += w[c] * g[c];
    globc[bt * CC + o] = s;
}

// ------ Kernel B2: pack Wp[:, 0:512] -> bf16 WMMA A-fragment order ------
// tile(mt,kt) = 32 lanes x 16 bf16.  lane (g=lane/16, m=lane%16):
//   e<8  -> k = kt*32 + 8*g + e
//   e>=8 -> k = kt*32 + 16 + 8*g + (e-8)
__global__ void k_pack(const float* __restrict__ Wp, unsigned short* __restrict__ wf) {
    const int flat = blockIdx.x * 256 + threadIdx.x;   // < 131072
    const int tile = flat >> 9, rem = flat & 511;
    const int lane = rem >> 4, e = rem & 15;
    const int mt = tile >> 4, kt = tile & 15;
    const int o = mt * 16 + (lane & 15);
    const int kb = (lane >> 4) * 8;
    const int kl = (e < 8) ? (kb + e) : (16 + kb + (e - 8));
    const int k = kt * 32 + kl;
    wf[flat] = f2bf(Wp[(size_t)o * (3 * CC) + k]);
}

// --------- Kernel C: fused WMMA GEMM + head MLPs + pair emission ---------
// block = (voxel tile of 64, bt); 8 waves; wave w owns rows o in [32w,32w+32)
__launch_bounds__(256)
__global__ void k_proj(const float* __restrict__ features, const float* __restrict__ pos,
                       const float* __restrict__ globc, const unsigned short* __restrict__ wfrag,
                       const float* __restrict__ Wo, const float* __restrict__ bo,
                       const float* __restrict__ Ww, const float* __restrict__ bw,
                       int* __restrict__ pairIdx, float* __restrict__ pairC) {
    extern __shared__ char smem[];
    unsigned short* sB = (unsigned short*)smem;              // 4nt*16kt*512 = 32768 ushorts (64KB)
    float* sGc = (float*)(smem + 65536);                     // 256 f
    float* sWo = (float*)(smem + 65536 + 1024);              // 96 f
    float* sWw = (float*)(smem + 65536 + 1024 + 384);        // 64 f

    const int vt = blockIdx.x, bt = blockIdx.y;
    const int v0 = vt * 64;
    const int b = bt / TT, t = bt % TT;
    const int tid = threadIdx.x;

    if (tid < 256) sGc[tid] = globc[bt * CC + tid];
    if (tid < 96)  sWo[tid] = Wo[tid];
    if (tid < 64)  sWw[tid] = Ww[tid];

    // Stage activations (rows 0..255 feat, 256..511 pos) as bf16 B-fragments.
    for (int it = 0; it < 32; ++it) {
        const int flat = tid + 256 * it;                     // 512 k * 16 v4
        const int k = flat >> 4, v4 = flat & 15, vloc = v4 * 4;
        const float* srcp = (k < CC) ? features : pos;
        const int c = (k < CC) ? k : (k - CC);
        const float4 x = *(const float4*)(srcp + (((size_t)(b * CC + c)) * TT + t) * VOX + v0 + vloc);
        const int kt = k >> 5, kq = k & 31;
        const int lgrp = (kq >> 3) & 1, e = (kq & 7) + ((kq >> 4) << 3);
        const float xv[4] = { x.x, x.y, x.z, x.w };
        #pragma unroll
        for (int i = 0; i < 4; ++i) {
            const int nloc = vloc + i, nt = nloc >> 4, np = nloc & 15;
            const int lane = np + 16 * lgrp;
            sB[(nt * 16 + kt) * 512 + lane * 16 + e] = f2bf(xv[i]);
        }
    }
    __syncthreads();

    const int w = tid >> 5, lane = tid & 31;
    const int mt0 = w * 2, mt1 = w * 2 + 1;
    v8f acc[2][4];
    const v8f z = { 0.f, 0.f, 0.f, 0.f, 0.f, 0.f, 0.f, 0.f };
    #pragma unroll
    for (int i = 0; i < 2; ++i)
        #pragma unroll
        for (int j = 0; j < 4; ++j) acc[i][j] = z;

    const v16bf* Ag = (const v16bf*)wfrag;
    const v16bf* Bs = (const v16bf*)sB;
    #pragma unroll
    for (int kt = 0; kt < 16; ++kt) {
        const v16bf a0 = Ag[(mt0 * 16 + kt) * 32 + lane];
        const v16bf a1 = Ag[(mt1 * 16 + kt) * 32 + lane];
        #pragma unroll
        for (int nt = 0; nt < 4; ++nt) {
            const v16bf bf = Bs[(nt * 16 + kt) * 32 + lane];
            acc[0][nt] = __builtin_amdgcn_wmma_f32_16x16x32_bf16(
                false, a0, false, bf, (short)0, acc[0][nt], false, false);
            acc[1][nt] = __builtin_amdgcn_wmma_f32_16x16x32_bf16(
                false, a1, false, bf, (short)0, acc[1][nt], false, false);
        }
    }

    // Epilogue: bias+ReLU, head-w MLPs (close with shfl_xor 16), emit pairs.
    const int g = lane >> 4, np = lane & 15;
    const float bo0 = bo[0], bo1 = bo[1], bo2 = bo[2];
    const float bw0 = bw[0], bw1 = bw[1];
    for (int nt = 0; nt < 4; ++nt) {
        float so0 = 0.f, so1 = 0.f, so2 = 0.f, sw0 = 0.f, sw1 = 0.f;
        #pragma unroll
        for (int mtj = 0; mtj < 2; ++mtj) {
            #pragma unroll
            for (int r = 0; r < 8; ++r) {
                const int d = mtj * 16 + 8 * g + r;
                const int o = w * 32 + d;
                float x = acc[mtj][nt][r] + sGc[o];
                x = fmaxf(x, 0.f);
                so0 += x * sWo[d * 3 + 0];
                so1 += x * sWo[d * 3 + 1];
                so2 += x * sWo[d * 3 + 2];
                sw0 += x * sWw[d * 2 + 0];
                sw1 += x * sWw[d * 2 + 1];
            }
        }
        so0 += __shfl_xor(so0, 16);
        so1 += __shfl_xor(so1, 16);
        so2 += __shfl_xor(so2, 16);
        sw0 += __shfl_xor(sw0, 16);
        sw1 += __shfl_xor(sw1, 16);

        const float off0 = 1.f / (1.f + __expf(-(so0 + bo0)));
        const float off1 = 1.f / (1.f + __expf(-(so1 + bo1)));
        const float off2 = 1.f / (1.f + __expf(-(so2 + bo2)));
        const float wg = 1.f / (1.f + __expf(-((sw1 + bw1) - (sw0 + bw0))));  // softmax[...,1]

        if (g == 0) {
            const int v = v0 + nt * 16 + np;
            const int h = v >> 7, y = (v >> 3) & 15, l = v & 7;
            const float basep[3] = { h * (1.f / 15.f), y * (1.f / 15.f), l * (1.f / 7.f) };
            const float sz[3] = { 16.f, 16.f, 8.f };
            const float offv[3] = { off0, off1, off2 };
            int lo[3]; float fr[3];
            #pragma unroll
            for (int ax = 0; ax < 3; ++ax) {
                const float p = fminf(fmaxf(basep[ax], EPSV), 1.f - EPSV);
                const float lg = __logf(p / (1.f - p));
                const float sp = 1.f / (1.f + __expf(-(lg + offv[ax])));
                const float pix = sp * sz[ax] - 0.5f;
                const float fl = floorf(pix);
                lo[ax] = (int)fl; fr[ax] = pix - fl;
            }
            const size_t pb = (((size_t)bt * VOX + v) * NOFF + w) * 8;
            int corner = 0;
            #pragma unroll
            for (int dz = 0; dz < 2; ++dz)
                #pragma unroll
                for (int dy = 0; dy < 2; ++dy)
                    #pragma unroll
                    for (int dx = 0; dx < 2; ++dx) {
                        const int iz = lo[0] + dz, iy = lo[1] + dy, ix = lo[2] + dx;
                        const bool valid = (iz >= 0) && (iz < 16) && (iy >= 0) && (iy < 16) &&
                                           (ix >= 0) && (ix < 8);
                        const float wz = dz ? fr[0] : 1.f - fr[0];
                        const float wy = dy ? fr[1] : 1.f - fr[1];
                        const float wx = dx ? fr[2] : 1.f - fr[2];
                        const int cz = iclamp(iz, 0, 15), cy = iclamp(iy, 0, 15), cx = iclamp(ix, 0, 7);
                        const int lin = (cz * 16 + cy) * 8 + cx;
                        pairIdx[pb + corner] = lin;
                        pairC[pb + corner] = valid ? (wz * wy * wx * wg) : 0.f;
                        ++corner;
                    }
        }
    }
}

// --------- Kernel D: async-DMA slab into LDS (transposed), gather, store ---------
// block = (channel slab of 32, bt). LDS: [2048 voxels][36 floats] = 288 KB.
// The global->LDS transpose ([c][v] -> padded [v][c]) is done by the async
// copy engine itself: per-lane scattered LDS destinations, ASYNCcnt-tracked.
__launch_bounds__(256)
__global__ void k_sample(const float* __restrict__ features,
                         const int* __restrict__ pairIdx, const float* __restrict__ pairC,
                         float* __restrict__ out) {
    extern __shared__ char smem[];
    float* sF = (float*)smem;
    const int slab = blockIdx.x, bt = blockIdx.y;
    const int b = bt / TT, t = bt % TT;
    const int tid = threadIdx.x;

    // Async global->LDS copy with transpose: element (c, v) lands at sF[v*36 + cl].
    for (int it = 0; it < 256; ++it) {
        const int flat = tid + 256 * it;                 // 32 c * 2048 v
        const int cl = flat >> 11, v = flat & 2047;
        const int c = slab * 32 + cl;
        const unsigned moff =
            (unsigned)(((((size_t)(b * CC + c)) * TT + t) * VOX + v) * sizeof(float));
        const unsigned loff = (unsigned)(size_t)(sF + (v * 36 + cl));   // LDS byte offset
        asm volatile("global_load_async_to_lds_b32 %0, %1, %2"
                     :: "v"(loff), "v"(moff), "s"(features)
                     : "memory");
    }
    wait_async0();
    __syncthreads();

    for (int i = 0; i < 8; ++i) {
        const int v = tid + 256 * i;
        float4 acc[8];
        #pragma unroll
        for (int q = 0; q < 8; ++q) acc[q] = make_float4(0.f, 0.f, 0.f, 0.f);
        const size_t pb = ((size_t)bt * VOX + v) * 64;
        const int* pi = pairIdx + pb;
        const float* pc = pairC + pb;
        __builtin_prefetch(pi, 0, 0);
        __builtin_prefetch(pc, 0, 0);
        for (int j = 0; j < 64; ++j) {
            const int idx = pi[j];
            const float cf = pc[j];
            const float4* row = (const float4*)(sF + idx * 36);   // 16B aligned (144*idx)
            #pragma unroll
            for (int q = 0; q < 8; ++q) {
                const float4 rv = row[q];
                acc[q].x += cf * rv.x; acc[q].y += cf * rv.y;
                acc[q].z += cf * rv.z; acc[q].w += cf * rv.w;
            }
        }
        #pragma unroll
        for (int q = 0; q < 8; ++q) {
            const float av[4] = { acc[q].x, acc[q].y, acc[q].z, acc[q].w };
            #pragma unroll
            for (int e = 0; e < 4; ++e) {
                const int cl = q * 4 + e;
                const int c = slab * 32 + cl;
                out[(((size_t)(b * CC + c)) * TT + t) * VOX + v] = sF[v * 36 + cl] + av[e];
            }
        }
    }
}

extern "C" void kernel_launch(void* const* d_in, const int* in_sizes, int n_in,
                              void* d_out, int out_size, void* d_ws, size_t ws_size,
                              hipStream_t stream) {
    const float* features = (const float*)d_in[0];
    const float* pos      = (const float*)d_in[1];
    const float* Wp       = (const float*)d_in[2];
    const float* bp       = (const float*)d_in[3];
    const float* Wo       = (const float*)d_in[4];
    const float* bo       = (const float*)d_in[5];
    const float* Ww       = (const float*)d_in[6];
    const float* bw       = (const float*)d_in[7];
    float* out = (float*)d_out;

    char* ws = (char*)d_ws;
    float* glob           = (float*)ws;                                 // 32 KB
    float* globc          = (float*)(ws + 32768);                       // 32 KB
    unsigned short* wfrag = (unsigned short*)(ws + 65536);              // 256 KB
    int* pairI            = (int*)(ws + 65536 + 262144);                // 16 MB
    float* pairCf         = (float*)(ws + 65536 + 262144 + 16777216);   // 16 MB

    k_mean<<<BTn * CC, 256, 0, stream>>>(features, glob);
    k_globc<<<BTn, 256, 0, stream>>>(Wp, bp, glob, globc);
    k_pack<<<512, 256, 0, stream>>>(Wp, wfrag);

    dim3 gc(VOX / 64, BTn);
    k_proj<<<gc, 256, 65536 + 1024 + 384 + 256, stream>>>(
        features, pos, globc, wfrag, Wo, bo, Ww, bw, pairI, pairCf);

    dim3 gd(CC / 32, BTn);
    k_sample<<<gd, 256, VOX * 36 * 4, stream>>>(features, pairI, pairCf, out);
}
